// MACTransformer_23218593202421
// MI455X (gfx1250) — matrix operand order repaired and checked by reading, not compiled, not run on hardware
//
#include <hip/hip_runtime.h>
#include <hip/hip_bf16.h>

// ---------------- model constants (match reference) ----------------
constexpr int kV   = 32000;
constexpr int kD   = 1024;
constexpr int kL   = 4;
constexpr int kH   = 16;
constexpr int kDH  = 64;
constexpr int kSEG = 512;
constexpr int kM   = 16;
constexpr int kP   = 4;
constexpr int kS   = 2048;
constexpr int kW   = kS / kSEG;     // 4 windows
constexpr int kN   = kSEG + kM;     // 528 tokens per window
constexpr int kDI  = 2730;          // int(1024*8/3)
constexpr int kR   = kW * kN;       // 2112 residual rows
constexpr int kNP  = kN + kP;       // 532 attention columns
constexpr int kNPp = 544;           // padded attention K (mult of 32)
constexpr int kDIp = 2752;          // padded DI (mult of 32)
constexpr int kDI2 = 2 * kDI;       // 5460
constexpr int kDI2p= 5472;          // padded ff1 cols (mult of 32)

typedef __attribute__((ext_vector_type(16))) __bf16 v16bf;
typedef __attribute__((ext_vector_type(8)))  float  v8f;

// ---------------- Tensor Data Mover availability ----------------
#if defined(__has_builtin)
#  if __has_builtin(__builtin_amdgcn_tensor_load_to_lds) && \
      __has_builtin(__builtin_amdgcn_s_wait_tensorcnt)
#    define HAVE_TDM 1
#  endif
#endif
#ifndef HAVE_TDM
#  define HAVE_TDM 0
#endif
#if HAVE_TDM
#  if __has_include(<hip/amd_detail/amd_gfx1250_TDM.h>)
#    define TDM_SIX_ARGS 1
#  else
#    define TDM_SIX_ARGS 0
#  endif
typedef __attribute__((ext_vector_type(4))) unsigned tdm_u32x4;
typedef __attribute__((ext_vector_type(8))) int      tdm_i32x8;
typedef __attribute__((ext_vector_type(4))) int      tdm_i32x4;

// 2D tile DMA: global (rowsValid x 32 bf16, row stride = strideElems) -> LDS
// [64 rows x 32 bf16] contiguous; rows beyond rowsValid are zero-filled by TDM.
static __device__ __forceinline__ void tdm_load_tile(unsigned ldsOff,
                                                     unsigned long long gaddr,
                                                     unsigned strideElems,
                                                     int rowsValid) {
    unsigned rv = (unsigned)rowsValid;
    tdm_u32x4 g0;
    g0[0] = 1u;                                                   // count=1, is_restore=0
    g0[1] = ldsOff;                                               // lds_addr (bytes)
    g0[2] = (unsigned)(gaddr & 0xFFFFFFFFull);                    // global_addr[31:0]
    g0[3] = (unsigned)((gaddr >> 32) & 0x1FFFFFFull) | (2u << 30);// addr[56:32] | type=2
    tdm_i32x8 g1;
    g1[0] = (int)(1u << 16);                                      // data_size=1 (2 bytes)
    g1[1] = (int)(32u << 16);                                     // tensor_dim0 = 32
    g1[2] = (int)((rv & 0xFFFFu) << 16);                          // tensor_dim1 lo16
    g1[3] = (int)(((rv >> 16) & 0xFFFFu) | (32u << 16));          // dim1 hi16 | tile_dim0=32
    g1[4] = 64;                                                   // tile_dim1 = 64
    g1[5] = (int)strideElems;                                     // tensor_dim0_stride lo32
    g1[6] = 0;
    g1[7] = 0;
    tdm_i32x4 gz = {0, 0, 0, 0};
#if TDM_SIX_ARGS
    tdm_i32x8 gz8 = {0, 0, 0, 0, 0, 0, 0, 0};
    __builtin_amdgcn_tensor_load_to_lds(g0, g1, gz, gz, gz8, 0);
#else
    __builtin_amdgcn_tensor_load_to_lds(g0, g1, gz, gz, 0);
#endif
}
#endif // HAVE_TDM

static __device__ __forceinline__ __bf16 f2bf(float f) {
    union { float f; unsigned u; } a; a.f = f;
    unsigned r = (a.u + 0x7FFFu + ((a.u >> 16) & 1u)) >> 16;   // RNE
    union { unsigned short s; __bf16 b; } o; o.s = (unsigned short)r;
    return o.b;
}
static __device__ __forceinline__ float bf2f(__bf16 b) {
    union { unsigned short s; __bf16 b; } i; i.b = b;
    union { unsigned u; float f; } o; o.u = ((unsigned)i.s) << 16;
    return o.f;
}

// ---------------- weight conversion (f32 -> bf16, padded) ----------------
__global__ __launch_bounds__(256)
void convcols_k(const float* __restrict__ src, __bf16* __restrict__ dst,
                long long rows, int cols, int dstStride) {
    long long i = (long long)blockIdx.x * 256 + threadIdx.x;
    long long tot = rows * dstStride;
    if (i >= tot) return;
    int c = (int)(i % dstStride); long long r = i / dstStride;
    dst[i] = (c < cols) ? f2bf(src[r * cols + c]) : f2bf(0.f);
}

__global__ __launch_bounds__(256)
void convrows_k(const float* __restrict__ src, __bf16* __restrict__ dst,
                int layers, int srcRows, int dstRows, int cols) {
    long long i = (long long)blockIdx.x * 256 + threadIdx.x;
    long long tot = (long long)layers * dstRows * cols;
    if (i >= tot) return;
    int c = (int)(i % cols); long long t = i / cols;
    int r = (int)(t % dstRows); int l = (int)(t / dstRows);
    dst[i] = (r < srcRows) ? f2bf(src[((long long)l * srcRows + r) * cols + c]) : f2bf(0.f);
}

// ---------------- embedding + positional ----------------
__global__ __launch_bounds__(256)
void embed_k(const int* __restrict__ ids, const float* __restrict__ tok,
             const float* __restrict__ pw, const float* __restrict__ ps,
             const float* __restrict__ mems, float* __restrict__ x) {
    long long i = (long long)blockIdx.x * 256 + threadIdx.x;
    long long tot = (long long)kW * kN * kD;
    if (i >= tot) return;
    int d = (int)(i % kD);
    long long t = i / kD;
    int n = (int)(t % kN);
    int w = (int)(t / kN);
    float v;
    if (n < kM) v = mems[(long long)n * kD + d];
    else        v = tok[(long long)ids[w * kSEG + (n - kM)] * kD + d];
    x[i] = v + pw[w * kD + d] + ps[n * kD + d];
}

// ---------------- rmsnorm: f32 in -> bf16 out ----------------
__global__ __launch_bounds__(256)
void rmsnorm_k(const float* __restrict__ in, const float* __restrict__ g,
               __bf16* __restrict__ out, int skipMems) {
    int row = blockIdx.x;
    long long ir = row;
    if (skipMems) { int w = row / kSEG, s = row % kSEG; ir = (long long)w * kN + kM + s; }
    const float* xr = in + ir * kD;
    __bf16* orow = out + (long long)row * kD;
    float ss = 0.f;
    for (int i = threadIdx.x; i < kD; i += 256) { float v = xr[i]; ss += v * v; }
    __shared__ float red[256];
    red[threadIdx.x] = ss; __syncthreads();
    for (int st = 128; st > 0; st >>= 1) {
        if (threadIdx.x < st) red[threadIdx.x] += red[threadIdx.x + st];
        __syncthreads();
    }
    float inv = rsqrtf(red[0] / (float)kD + 1e-6f);
    for (int i = threadIdx.x; i < kD; i += 256) orow[i] = f2bf(xr[i] * inv * g[i]);
}

// ---------------- generic strided BF16-WMMA GEMM ----------------
// C[m,n] (+)= act(scale * sum_k A[m,k]*B[k,n] + bias[n]);  K % 32 == 0 required.
// 64x64 block tile, 8 wave32 waves, each wave owns a 16x32 strip (2 wmma accumulators).
// Double-buffered LDS tiles; A (and K-contiguous B) fetched by the Tensor Data Mover
// overlapped with WMMA; single barrier per K-step.
template <int BKC>
__global__ __launch_bounds__(256)
void gemm_bf16_k(const __bf16* __restrict__ A, const __bf16* __restrict__ B,
                 const float* __restrict__ bias, float* __restrict__ C,
                 __bf16* __restrict__ Cb,
                 int M, int Nd, int NdB, int K,
                 long long aOff, long long aSW, long long aSH, int aRow,
                 long long bOff, long long bSW, long long bSH, int bRow, int bCol,
                 long long cOff, long long cSW, long long cSH, int cRow,
                 int zDivH, float scale, int accum, int actSigmoid) {
    int z = blockIdx.z;
    int wz = z / zDivH, hz = z % zDivH;
    const __bf16* Ap = A + aOff + (long long)wz * aSW + (long long)hz * aSH;
    const __bf16* Bp = B + bOff + (long long)wz * bSW + (long long)hz * bSH;
    long long cBase = cOff + (long long)wz * cSW + (long long)hz * cSH;

    __shared__ __align__(16) __bf16 lsA[2][64 * 32];   // [m][k], double-buffered
    __shared__ __align__(16) __bf16 lsB[2][64 * 32];   // [n][k], double-buffered

    int tid  = threadIdx.x;
    int lane = tid & 31;
    int wv   = tid >> 5;
    int sm   = wv & 3;
    int sn   = wv >> 2;

    int m0 = blockIdx.y * 64;
    int n0 = blockIdx.x * 64;

    v8f c0 = {}; v8f c1 = {};

#if HAVE_TDM
    unsigned ldsAoff = (unsigned)(unsigned long long)(void*)&lsA[0][0];
    unsigned ldsBoff = (unsigned)(unsigned long long)(void*)&lsB[0][0];
    unsigned long long aBase =
        (unsigned long long)(const void*)Ap + 2ull * (unsigned long long)((long long)m0 * aRow);
    unsigned long long bBase =
        (unsigned long long)(const void*)Bp + 2ull * (unsigned long long)((long long)n0 * bCol);
#else
    int acm = tid >> 2,  ack = (tid & 3) << 3;     // A: 64 rows x 4 chunks
    int bcn = tid >> 2,  bck = (tid & 3) << 3;     // B k-contig: 64 n-rows x 4 chunks
#endif
    int bck2 = tid >> 3, bcn2 = (tid & 7) << 3;    // B n-contig: 32 k-rows x 8 chunks

    int nK = K >> 5;

    // start loads of tile kt into LDS buffer p
    auto issueTile = [&](int kt, int p) {
        int k0 = kt << 5;
#if HAVE_TDM
        if (wv == 0)
            tdm_load_tile(ldsAoff + (unsigned)p * 4096u,
                          aBase + 2ull * (unsigned)k0, (unsigned)aRow, M - m0);
        if (BKC) {
            if (wv == 1)
                tdm_load_tile(ldsBoff + (unsigned)p * 4096u,
                              bBase + 2ull * (unsigned)k0, (unsigned)bCol, NdB - n0);
        } else {
            int gk = k0 + bck2, gn = n0 + bcn2;
            union { uint4 q; __bf16 e[8]; } vb;
            vb.q = (uint4){0u, 0u, 0u, 0u};
            if (gn + 7 < NdB) vb.q = *reinterpret_cast<const uint4*>(Bp + (long long)gk * bRow + gn);
            #pragma unroll
            for (int j = 0; j < 8; ++j) lsB[p][(bcn2 + j) * 32 + bck2] = vb.e[j];
        }
#else
        {
            int gm = m0 + acm, gk = k0 + ack;
            uint4 va = {0u, 0u, 0u, 0u};
            if (gm < M) va = *reinterpret_cast<const uint4*>(Ap + (long long)gm * aRow + gk);
            *reinterpret_cast<uint4*>(&lsA[p][acm * 32 + ack]) = va;
        }
        if (BKC) {
            int gn = n0 + bcn, gk = k0 + bck;
            uint4 vb = {0u, 0u, 0u, 0u};
            if (gn < NdB) vb = *reinterpret_cast<const uint4*>(Bp + (long long)gn * bCol + gk);
            *reinterpret_cast<uint4*>(&lsB[p][bcn * 32 + bck]) = vb;
        } else {
            int gk = k0 + bck2, gn = n0 + bcn2;
            union { uint4 q; __bf16 e[8]; } vb;
            vb.q = (uint4){0u, 0u, 0u, 0u};
            if (gn + 7 < NdB) vb.q = *reinterpret_cast<const uint4*>(Bp + (long long)gk * bRow + gn);
            #pragma unroll
            for (int j = 0; j < 8; ++j) lsB[p][(bcn2 + j) * 32 + bck2] = vb.e[j];
        }
#endif
    };

    auto compute = [&](int p) {
        union { uint4 q[2]; v16bf v; } fa, fb0, fb1;
        int am = sm * 16 + (lane & 15);
        int ao = (lane >> 4) * 16;
        fa.q[0] = *reinterpret_cast<const uint4*>(&lsA[p][am * 32 + ao]);
        fa.q[1] = *reinterpret_cast<const uint4*>(&lsA[p][am * 32 + ao + 8]);
        int bn0 = (sn * 2 + 0) * 16 + (lane & 15);
        int bn1 = (sn * 2 + 1) * 16 + (lane & 15);
        int bo  = (lane >> 4) * 16;
        fb0.q[0] = *reinterpret_cast<const uint4*>(&lsB[p][bn0 * 32 + bo]);
        fb0.q[1] = *reinterpret_cast<const uint4*>(&lsB[p][bn0 * 32 + bo + 8]);
        fb1.q[0] = *reinterpret_cast<const uint4*>(&lsB[p][bn1 * 32 + bo]);
        fb1.q[1] = *reinterpret_cast<const uint4*>(&lsB[p][bn1 * 32 + bo + 8]);
        c0 = __builtin_amdgcn_wmma_f32_16x16x32_bf16(false, fa.v, false, fb0.v,
                                                     (short)0, c0, false, false);
        c1 = __builtin_amdgcn_wmma_f32_16x16x32_bf16(false, fa.v, false, fb1.v,
                                                     (short)0, c1, false, false);
    };

    // pipelined: wait(kt) -> barrier -> issue(kt+1, other buf) -> compute(kt)
    auto step = [&](int kt, int p) {
#if HAVE_TDM
        __builtin_amdgcn_s_wait_tensorcnt(0);   // no-op for waves with TENSORcnt==0
#endif
        __syncthreads();
        if (kt + 1 < nK) issueTile(kt + 1, p ^ 1);
        compute(p);
    };

    issueTile(0, 0);
    int kt = 0;
    while (kt + 2 <= nK) { step(kt, 0); step(kt + 1, 1); kt += 2; }
    if (kt < nK) step(kt, 0);   // pairs preserve parity: tail uses buffer 0

    int col0 = n0 + sn * 32 + (lane & 15);
    int col1 = col0 + 16;
    int mrow = m0 + sm * 16 + ((lane >> 4) * 8);
    #pragma unroll
    for (int r = 0; r < 8; ++r) {
        int gm = mrow + r;
        if (gm >= M) continue;
        float v0 = c0[r] * scale;
        float v1 = c1[r] * scale;
        if (col0 < Nd) {
            if (bias) v0 += bias[col0];
            if (actSigmoid) v0 = 1.f / (1.f + __expf(-v0));
            long long ci = cBase + (long long)gm * cRow + col0;
            if (C) { if (accum) C[ci] += v0; else C[ci] = v0; }
            if (Cb) Cb[ci] = f2bf(v0);
        }
        if (col1 < Nd) {
            if (bias) v1 += bias[col1];
            if (actSigmoid) v1 = 1.f / (1.f + __expf(-v1));
            long long ci = cBase + (long long)gm * cRow + col1;
            if (C) { if (accum) C[ci] += v1; else C[ci] = v1; }
            if (Cb) Cb[ci] = f2bf(v1);
        }
    }
}

// ---------------- small elementwise kernels ----------------
__global__ __launch_bounds__(256)
void copy_vfirst_k(const float* __restrict__ qkv, float* __restrict__ vf) {
    long long i = (long long)blockIdx.x * 256 + threadIdx.x;
    if (i >= (long long)kR * kD) return;
    int d = (int)(i % kD); long long r = i / kD;
    vf[i] = qkv[r * (3 * kD) + 2 * kD + d];
}

__global__ __launch_bounds__(256)
void vmix_apply_k(float* __restrict__ qkv, const float* __restrict__ vf,
                  const float* __restrict__ mix) {
    long long i = (long long)blockIdx.x * 256 + threadIdx.x;
    if (i >= (long long)kR * kD) return;
    int d = (int)(i % kD); long long r = i / kD;
    int h = d >> 6;
    float m = mix[r * kH + h];
    long long vi = r * (3 * kD) + 2 * kD + d;
    qkv[vi] = qkv[vi] * m + vf[i] * (1.f - m);
}

// RoPE: read f32 q/k from qkv, write roped bf16 into head-major qb/kb [w][h][n][64]
__global__ __launch_bounds__(256)
void rope_k(const float* __restrict__ qkv, __bf16* __restrict__ qb, __bf16* __restrict__ kb) {
    long long i = (long long)blockIdx.x * 256 + threadIdx.x;
    long long tot = (long long)kR * kH * (kDH / 2) * 2;
    if (i >= tot) return;
    int qk = (int)(i & 1); long long t = i >> 1;
    int p = (int)(t % (kDH / 2)); t /= (kDH / 2);
    int h = (int)(t % kH); long long r = t / kH;
    int w = (int)(r / kN);
    int pos = (int)(r % kN);
    float inv = __powf(10000.f, -(float)(2 * p) / (float)kDH);
    float f = (float)pos * inv;
    float cs = __cosf(f), sn = __sinf(f);
    long long src = r * (3 * kD) + (long long)qk * kD + h * kDH + 2 * p;
    float x1 = qkv[src], x2 = qkv[src + 1];
    __bf16* dst = (qk == 0 ? qb : kb) + (((long long)(w * kH + h) * kN) + pos) * kDH + 2 * p;
    dst[0] = f2bf(x1 * cs - x2 * sn);
    dst[1] = f2bf(x1 * sn + x2 * cs);
}

// persistent-slot scores: scores[w,h,n,:P] = scale * q . persist_mem[l,0]
__global__ __launch_bounds__(256)
void sp_k(const __bf16* __restrict__ qb, const float* __restrict__ pm0,
          float* __restrict__ scores) {
    long long i = (long long)blockIdx.x * 256 + threadIdx.x;
    long long tot = (long long)kW * kH * kN * kP;
    if (i >= tot) return;
    int p = (int)(i % kP); long long t = i / kP;
    int n = (int)(t % kN); t /= kN;
    int h = (int)(t % kH); int w = (int)(t / kH);
    const __bf16* q = qb + (((long long)(w * kH + h) * kN) + n) * kDH;
    const float* mm = pm0 + ((long long)h * kP + p) * kDH;
    float s = 0.f;
    #pragma unroll 8
    for (int d = 0; d < kDH; ++d) s += bf2f(q[d]) * mm[d];
    scores[((long long)(w * kH + h) * kN + n) * kNP + p] = s * 0.125f;
}

// causal softmax: read f32 scores row (stride kNP), write bf16 probs row (stride kNPp)
__global__ __launch_bounds__(256)
void softmax_k(float* __restrict__ scores, __bf16* __restrict__ pb) {
    long long row = blockIdx.x;          // over W*H*N rows
    int n = (int)(row % kN);
    float* s = scores + row * kNP;
    __bf16* p = pb + row * kNPp;
    int valid = kP + n + 1;
    __shared__ float red[256];
    float mx = -1e30f;
    for (int i = threadIdx.x; i < valid; i += 256) mx = fmaxf(mx, s[i]);
    red[threadIdx.x] = mx; __syncthreads();
    for (int st = 128; st > 0; st >>= 1) {
        if (threadIdx.x < st) red[threadIdx.x] = fmaxf(red[threadIdx.x], red[threadIdx.x + st]);
        __syncthreads();
    }
    mx = red[0]; __syncthreads();
    float sum = 0.f;
    for (int i = threadIdx.x; i < valid; i += 256) {
        float e = __expf(s[i] - mx); s[i] = e; sum += e;
    }
    red[threadIdx.x] = sum; __syncthreads();
    for (int st = 128; st > 0; st >>= 1) {
        if (threadIdx.x < st) red[threadIdx.x] += red[threadIdx.x + st];
        __syncthreads();
    }
    float inv = 1.f / red[0];
    for (int i = threadIdx.x; i < valid; i += 256) p[i] = f2bf(s[i] * inv);
    for (int i = valid + threadIdx.x; i < kNPp; i += 256) p[i] = f2bf(0.f);
}

// Vcat bf16: [w][h][j<kNPp][64]; j<P persistent, P<=j<NP value rows, rest zero
__global__ __launch_bounds__(256)
void vcat_k(const float* __restrict__ qkv, const float* __restrict__ pm1,
            __bf16* __restrict__ vc) {
    long long i = (long long)blockIdx.x * 256 + threadIdx.x;
    long long tot = (long long)kW * kH * kNPp * kDH;
    if (i >= tot) return;
    int d = (int)(i & 63); long long t = i >> 6;
    int j = (int)(t % kNPp); t /= kNPp;
    int h = (int)(t % kH); int w = (int)(t / kH);
    float v = 0.f;
    if (j < kP) v = pm1[((long long)h * kP + j) * kDH + d];
    else if (j < kNP)
        v = qkv[(long long)(w * kN + (j - kP)) * (3 * kD) + 2 * kD + h * kDH + d];
    vc[i] = f2bf(v);
}

// SiLU-gate: ffh_bf16[r][c<kDIp] = silu(g1)*a1 (zero pad c>=kDI)
__global__ __launch_bounds__(256)
void silu_mul_k(const float* __restrict__ u, __bf16* __restrict__ out) {
    long long i = (long long)blockIdx.x * 256 + threadIdx.x;
    if (i >= (long long)kR * kDIp) return;
    int c = (int)(i % kDIp); long long r = i / kDIp;
    float v = 0.f;
    if (c < kDI) {
        float a1 = u[r * kDI2 + c];
        float g1 = u[r * kDI2 + kDI + c];
        v = a1 * (g1 / (1.f + __expf(-g1)));
    }
    out[i] = f2bf(v);
}

// ---------------- host orchestration ----------------
static inline int ceil_div(long long a, int b) { return (int)((a + b - 1) / b); }

extern "C" void kernel_launch(void* const* d_in, const int* in_sizes, int n_in,
                              void* d_out, int out_size, void* d_ws, size_t ws_size,
                              hipStream_t stream) {
    (void)in_sizes; (void)n_in; (void)out_size; (void)ws_size;
    const int*   ids      = (const int*)  d_in[0];
    const float* tok_emb  = (const float*)d_in[1];
    const float* pos_win  = (const float*)d_in[2];
    const float* pos_seg  = (const float*)d_in[3];
    const float* lt_mems  = (const float*)d_in[4];
    const float* attn_ng  = (const float*)d_in[5];
    const float* Wqkv     = (const float*)d_in[6];
    const float* bqkv     = (const float*)d_in[7];
    const float* Wo       = (const float*)d_in[8];
    const float* bo       = (const float*)d_in[9];
    const float* pmem     = (const float*)d_in[10];
    const float* vmix_w   = (const float*)d_in[11];
    const float* vmix_b   = (const float*)d_in[12];
    const float* ff_ng    = (const float*)d_in[13];
    const float* ff_w1    = (const float*)d_in[14];
    const float* ff_b1    = (const float*)d_in[15];
    const float* ff_w2    = (const float*)d_in[16];
    const float* ff_b2    = (const float*)d_in[17];
    const float* fin_ng   = (const float*)d_in[18];
    const float* W_logits = (const float*)d_in[19];
    float* out = (float*)d_out;

    float* ws = (float*)d_ws;
    long long o = 0;
    auto allocF = [&](long long n) { float* p = ws + o; o += (n + 15) & ~15LL; return p; };
    auto allocB = [&](long long nh) { __bf16* p = (__bf16*)(ws + o); o += (((nh + 1) / 2) + 15) & ~15LL; return p; };

    float* x       = allocF((long long)kR * kD);
    float* qkv     = allocF((long long)kR * 3 * kD);
    float* vfirst  = allocF((long long)kR * kD);
    float* mix     = allocF((long long)kR * kH);
    float* scoresF = allocF((long long)kW * kH * kN * kNP);   // also reused as ffu
    float* ffu     = scoresF;                                  // disjoint lifetime
    __bf16* hb     = allocB((long long)kR * kD);
    __bf16* qb     = allocB((long long)kR * kD);
    __bf16* kb     = allocB((long long)kR * kD);
    __bf16* pb     = allocB((long long)kW * kH * kN * kNPp);
    __bf16* vcb    = allocB((long long)kW * kH * kNPp * kDH);
    __bf16* attnb  = allocB((long long)kR * kD);
    __bf16* ffhb   = allocB((long long)kR * kDIp);
    __bf16* hfinb  = allocB((long long)kS * kD);
    __bf16* wqkvb  = allocB((long long)kL * kD * 3 * kD);
    __bf16* wob    = allocB((long long)kL * kD * kD);
    __bf16* w1b    = allocB((long long)kL * kD * kDI2p);
    __bf16* w2b    = allocB((long long)kL * kDIp * kD);
    __bf16* wlogb  = allocB((long long)kD * kV);
    __bf16* vmixb  = allocB((long long)kL * kD * kH);

    // ---- weight conversions (bf16, padded) ----
    convcols_k<<<ceil_div((long long)kL * kD * 3 * kD, 256), 256, 0, stream>>>(
        Wqkv, wqkvb, (long long)kL * kD, 3 * kD, 3 * kD);
    convcols_k<<<ceil_div((long long)kL * kD * kD, 256), 256, 0, stream>>>(
        Wo, wob, (long long)kL * kD, kD, kD);
    convcols_k<<<ceil_div((long long)kL * kD * kDI2p, 256), 256, 0, stream>>>(
        ff_w1, w1b, (long long)kL * kD, kDI2, kDI2p);
    convrows_k<<<ceil_div((long long)kL * kDIp * kD, 256), 256, 0, stream>>>(
        ff_w2, w2b, kL, kDI, kDIp, kD);
    convcols_k<<<ceil_div((long long)kD * kV, 256), 256, 0, stream>>>(
        W_logits, wlogb, (long long)kD, kV, kV);
    convcols_k<<<ceil_div((long long)kL * kD * kH, 256), 256, 0, stream>>>(
        vmix_w, vmixb, (long long)kL * kD, kH, kH);

    auto gemm = [&](const __bf16* A, const __bf16* B, const float* bias,
                    float* C, __bf16* Cb,
                    int M, int Nd, int NdB, int K,
                    long long aOff, long long aSW, long long aSH, int aRow,
                    long long bOff, long long bSW, long long bSH, int bRow, int bCol, int bKC,
                    long long cOff, long long cSW, long long cSH, int cRow,
                    int batch, int zDivH, float scale, int accum, int sig) {
        dim3 grid((Nd + 63) / 64, (M + 63) / 64, batch);
        if (bKC)
            gemm_bf16_k<1><<<grid, 256, 0, stream>>>(A, B, bias, C, Cb, M, Nd, NdB, K,
                                                     aOff, aSW, aSH, aRow,
                                                     bOff, bSW, bSH, bRow, bCol,
                                                     cOff, cSW, cSH, cRow,
                                                     zDivH, scale, accum, sig);
        else
            gemm_bf16_k<0><<<grid, 256, 0, stream>>>(A, B, bias, C, Cb, M, Nd, NdB, K,
                                                     aOff, aSW, aSH, aRow,
                                                     bOff, bSW, bSH, bRow, bCol,
                                                     cOff, cSW, cSH, cRow,
                                                     zDivH, scale, accum, sig);
    };

    embed_k<<<ceil_div((long long)kR * kD, 256), 256, 0, stream>>>(
        ids, tok_emb, pos_win, pos_seg, lt_mems, x);

    for (int l = 0; l < kL; ++l) {
        const float* pm0 = pmem + ((long long)(l * 2 + 0) * kH * kP * kDH);
        const float* pm1 = pmem + ((long long)(l * 2 + 1) * kH * kP * kDH);

        rmsnorm_k<<<kR, 256, 0, stream>>>(x, attn_ng + (long long)l * kD, hb, 0);

        // qkv = hb @ Wqkv[l] + bqkv[l]  (f32 out)
        gemm(hb, wqkvb + (long long)l * kD * 3 * kD, bqkv + (long long)l * 3 * kD, qkv, nullptr,
             kR, 3 * kD, 3 * kD, kD,
             0, 0, 0, kD,
             0, 0, 0, 3 * kD, 1, 0,
             0, 0, 0, 3 * kD,
             1, 1, 1.f, 0, 0);

        if (l == 0) {
            copy_vfirst_k<<<ceil_div((long long)kR * kD, 256), 256, 0, stream>>>(qkv, vfirst);
        } else {
            gemm(hb, vmixb + (long long)l * kD * kH, vmix_b + (long long)l * kH, mix, nullptr,
                 kR, kH, kH, kD,
                 0, 0, 0, kD,
                 0, 0, 0, kH, 1, 0,
                 0, 0, 0, kH,
                 1, 1, 1.f, 0, 1);
            vmix_apply_k<<<ceil_div((long long)kR * kD, 256), 256, 0, stream>>>(qkv, vfirst, mix);
        }

        rope_k<<<ceil_div((long long)kR * kH * (kDH / 2) * 2, 256), 256, 0, stream>>>(qkv, qb, kb);

        // scores[w,h,n,P:] = 0.125 * q . k   (z = w*H + h; B is K-contiguous -> TDM path)
        gemm(qb, kb, nullptr, scoresF, nullptr,
             kN, kN, kN, kDH,
             0, (long long)kH * kN * kDH, (long long)kN * kDH, kDH,
             0, (long long)kH * kN * kDH, (long long)kN * kDH, 1, kDH, 1,
             kP, (long long)kH * kN * kNP, (long long)kN * kNP, kNP,
             kW * kH, kH, 0.125f, 0, 0);

        sp_k<<<ceil_div((long long)kW * kH * kN * kP, 256), 256, 0, stream>>>(qb, pm0, scoresF);

        softmax_k<<<kW * kH * kN, 256, 0, stream>>>(scoresF, pb);

        vcat_k<<<ceil_div((long long)kW * kH * kNPp * kDH, 256), 256, 0, stream>>>(qkv, pm1, vcb);

        // attnb[w,n,h,:] = probs @ Vcat  (bf16 out only)
        gemm(pb, vcb, nullptr, nullptr, attnb,
             kN, kDH, kDH, kNPp,
             0, (long long)kH * kN * kNPp, (long long)kN * kNPp, kNPp,
             0, (long long)kH * kNPp * kDH, (long long)kNPp * kDH, kDH, 1, 0,
             0, (long long)kN * kD, (long long)kDH, kD,
             kW * kH, kH, 1.f, 0, 0);

        // x += attnb @ Wo[l] + bo[l]
        gemm(attnb, wob + (long long)l * kD * kD, bo + (long long)l * kD, x, nullptr,
             kR, kD, kD, kD,
             0, 0, 0, kD,
             0, 0, 0, kD, 1, 0,
             0, 0, 0, kD,
             1, 1, 1.f, 1, 0);

        rmsnorm_k<<<kR, 256, 0, stream>>>(x, ff_ng + (long long)l * kD, hb, 0);

        // ffu = hb @ ff_w1[l] + b1  (f32; B cols padded to kDI2p)
        gemm(hb, w1b + (long long)l * kD * kDI2p, ff_b1 + (long long)l * kDI2, ffu, nullptr,
             kR, kDI2, kDI2p, kD,
             0, 0, 0, kD,
             0, 0, 0, kDI2p, 1, 0,
             0, 0, 0, kDI2,
             1, 1, 1.f, 0, 0);

        silu_mul_k<<<ceil_div((long long)kR * kDIp, 256), 256, 0, stream>>>(ffu, ffhb);

        // x += ffhb @ ff_w2[l] + b2  (K padded to kDIp; w2b row-padded with zeros)
        gemm(ffhb, w2b + (long long)l * kDIp * kD, ff_b2 + (long long)l * kD, x, nullptr,
             kR, kD, kD, kDIp,
             0, 0, 0, kDIp,
             0, 0, 0, kD, 1, 0,
             0, 0, 0, kD,
             1, 1, 1.f, 1, 0);
    }

    rmsnorm_k<<<kS, 256, 0, stream>>>(x, fin_ng, hfinb, 1);

    // logits = hfinb @ W_logits  (f32 out to d_out)
    gemm(hfinb, wlogb, nullptr, out, nullptr,
         kS, kV, kV, kD,
         0, 0, 0, kD,
         0, 0, 0, kV, 1, 0,
         0, 0, 0, kV,
         1, 1, 1.f, 0, 0);
}
